// BasicBlockOurIn_61624190763759
// MI455X (gfx1250) — matrix operand order repaired and checked by reading, not compiled, not run on hardware
//
#include <hip/hip_runtime.h>
#include <hip/hip_bf16.h>
#include <stdint.h>

typedef __attribute__((ext_vector_type(16))) __bf16 v16bf;
typedef __attribute__((ext_vector_type(8)))  __bf16 v8bf;
typedef __attribute__((ext_vector_type(8)))  float  v8f;

#define NPTS  65536
#define CH    128
#define KVOL  27
#define NBATCH 8
#define PER   8192
#define NEG_SLOPE 0.01f
#define EPSV  1e-6f

// ---------------------------------------------------------------------------
// CDNA5 async global->LDS copy (ASYNCcnt-tracked), GV addressing mode.
// vdst = LDS byte offset VGPR, vaddr = 64-bit global address.
// ---------------------------------------------------------------------------
__device__ __forceinline__ void async_copy_b128(uint32_t lds_off, uint64_t gaddr) {
    asm volatile("global_load_async_to_lds_b128 %0, %1, off"
                 :: "v"(lds_off), "v"(gaddr) : "memory");
}
__device__ __forceinline__ void wait_asynccnt0() {
    asm volatile("s_wait_asynccnt 0x0" ::: "memory");
}

// ---------------------------------------------------------------------------
// fp32 -> bf16 elementwise convert
// ---------------------------------------------------------------------------
__global__ void cvt_bf16_kernel(const float* __restrict__ src,
                                __bf16* __restrict__ dst, int n) {
    int g = blockIdx.x * blockDim.x + threadIdx.x;
    if (g < n) dst[g] = (__bf16)src[g];
}

// ---------------------------------------------------------------------------
// Repack W[k][cin][cout] (f32) into WMMA-B bf16 tiles:
// dst[k][cc][nt][lane*16 + hh] = W[k][cc*32 + K][nt*16 + (lane&15)]
//   K = (hh&7) + (lane>=16 ? 8 : 0) + (hh>=8 ? 16 : 0)   (CDNA5 16-bit B 32x16 VGPR layout)
// ---------------------------------------------------------------------------
__global__ void repack_w_kernel(const float* __restrict__ w,
                                __bf16* __restrict__ dst) {
    int t = blockIdx.x * blockDim.x + threadIdx.x;
    if (t >= KVOL * 4 * 8 * 512) return;
    int hh   = t & 15;
    int lane = (t >> 4) & 31;
    int nt   = (t >> 9) & 7;
    int cc   = (t >> 12) & 3;
    int k    = t >> 14;
    int K    = (hh & 7) + ((lane >= 16) ? 8 : 0) + ((hh >= 8) ? 16 : 0);
    int cin  = cc * 32 + K;
    int cout = nt * 16 + (lane & 15);
    dst[t] = (__bf16)w[((size_t)k * CH + cin) * CH + cout];
}

// Gathered A fragment: 16x32 bf16 tile row for this lane (two 16B loads).
__device__ __forceinline__ v16bf load_a(const __bf16* __restrict__ fin,
                                        int idx, int cc, int hisel) {
    v16bf a;
    if (idx >= 0) {
        const __bf16* ap = fin + (size_t)idx * CH + cc * 32 + hisel * 8;
        v8bf lo = *(const v8bf*)ap;
        v8bf hi = *(const v8bf*)(ap + 16);
        #pragma unroll
        for (int i = 0; i < 8; ++i) { a[i] = lo[i]; a[8 + i] = hi[i]; }
    } else {
        #pragma unroll
        for (int i = 0; i < 16; ++i) a[i] = (__bf16)0.0f;
    }
    return a;
}

// ---------------------------------------------------------------------------
// Sparse conv as gather-GEMM with bf16 WMMA, f32 accumulate.
// Block = 256 threads = 8 waves; each wave computes 32 rows x 128 cols
// (two A fragments share every B fragment -> 2 WMMAs per LDS B read).
// Weights double-buffered in LDS via async global->LDS copies: slice k+1
// streams in while slice k is consumed by the matrix pipe.
// ---------------------------------------------------------------------------
__global__ __launch_bounds__(256) void sparse_conv_wmma_kernel(
    const __bf16* __restrict__ fin,   // [NPTS, CH] bf16
    const __bf16* __restrict__ pw,    // [KVOL, 4, 8, 512] packed B tiles
    const int*    __restrict__ nbr,   // [KVOL, NPTS]
    float*        __restrict__ fout)  // [NPTS, CH] f32
{
    __shared__ uint4 smem4[2][2048];  // 2 x 32 KB double-buffered weight slices

    const int tid      = threadIdx.x;
    const int lane     = tid & 31;
    const int wv       = tid >> 5;
    const int base_row = blockIdx.x * 256 + wv * 32;
    const int colsel   = lane & 15;   // N index within 16-wide tile
    const int hisel    = lane >> 4;   // 0 for lanes 0-15, 1 for 16-31

    v8f acc0[8] = {};
    v8f acc1[8] = {};

    const uint32_t lds_base = (uint32_t)(uintptr_t)(&smem4[0][0]);

    // Prologue: async-stage k=0 weight slice into buffer 0.
    {
        const uint4* src = (const uint4*)pw;
        #pragma unroll
        for (int i = 0; i < 8; ++i)
            async_copy_b128(lds_base + (uint32_t)(tid + i * 256) * 16u,
                            (uint64_t)(uintptr_t)(src + tid + i * 256));
    }

    for (int k = 0; k < KVOL; ++k) {
        const int cur = k & 1;
        wait_asynccnt0();     // this wave's copies for buf[cur] complete
        __syncthreads();      // all waves' copies visible; all done with k-1

        // Prefetch slice k+1 into the other buffer (overlaps with WMMAs).
        if (k + 1 < KVOL) {
            const uint4* src = (const uint4*)(pw + (size_t)(k + 1) * 16384);
            const uint32_t lbase = lds_base + (uint32_t)(cur ^ 1) * 32768u;
            #pragma unroll
            for (int i = 0; i < 8; ++i)
                async_copy_b128(lbase + (uint32_t)(tid + i * 256) * 16u,
                                (uint64_t)(uintptr_t)(src + tid + i * 256));
        }

        const __bf16* smem = (const __bf16*)(&smem4[cur][0]);
        const int idx0 = nbr[(size_t)k * NPTS + base_row + colsel];
        const int idx1 = nbr[(size_t)k * NPTS + base_row + 16 + colsel];

        #pragma unroll
        for (int cc = 0; cc < 4; ++cc) {
            v16bf a0 = load_a(fin, idx0, cc, hisel);
            v16bf a1 = load_a(fin, idx1, cc, hisel);
            #pragma unroll
            for (int nt = 0; nt < 8; ++nt) {
                v16bf b = *(const v16bf*)&smem[(cc * 8 + nt) * 512 + lane * 16];
                acc0[nt] = __builtin_amdgcn_wmma_f32_16x16x32_bf16(
                    false, a0, false, b, (short)0, acc0[nt], false, false);
                acc1[nt] = __builtin_amdgcn_wmma_f32_16x16x32_bf16(
                    false, a1, false, b, (short)0, acc1[nt], false, false);
            }
        }
    }

    // D layout: VGPR r -> M = r + hisel*8, N = lane&15
    const int mrow0 = base_row + hisel * 8;
    const int mrow1 = base_row + 16 + hisel * 8;
    #pragma unroll
    for (int nt = 0; nt < 8; ++nt) {
        #pragma unroll
        for (int r = 0; r < 8; ++r) {
            fout[(size_t)(mrow0 + r) * CH + nt * 16 + colsel] = acc0[nt][r];
            fout[(size_t)(mrow1 + r) * CH + nt * 16 + colsel] = acc1[nt][r];
        }
    }
}

// ---------------------------------------------------------------------------
// Instance-norm stats, stage 1: per (batch, slice of 256 rows) partial sums.
// Batches are contiguous 8192-row blocks (keys sorted in reference).
// ---------------------------------------------------------------------------
__global__ __launch_bounds__(256) void stats_partial_kernel(
    const float* __restrict__ x, float* __restrict__ psum,
    float* __restrict__ psq)
{
    const int b   = blockIdx.x >> 5;
    const int s   = blockIdx.x & 31;
    const int tid = threadIdx.x;
    const int c   = tid & 127;
    const int h   = tid >> 7;
    const size_t base = (size_t)b * PER + (size_t)s * 256;

    float sum = 0.f, sq = 0.f;
    for (int r = h; r < 256; r += 2) {
        float v = x[(base + r) * CH + c];
        sum += v; sq += v * v;
    }
    __shared__ float ss[256], sb[256];
    ss[tid] = sum; sb[tid] = sq;
    __syncthreads();
    if (tid < 128) {
        psum[(size_t)blockIdx.x * 128 + tid] = ss[tid] + ss[tid + 128];
        psq [(size_t)blockIdx.x * 128 + tid] = sb[tid] + sb[tid + 128];
    }
}

// Stage 2: reduce 32 slices -> mean / rstd per (batch, channel).
__global__ void stats_final_kernel(const float* __restrict__ psum,
                                   const float* __restrict__ psq,
                                   float* __restrict__ mean,
                                   float* __restrict__ rstd)
{
    const int b = blockIdx.x;
    const int c = threadIdx.x;   // 128 threads
    float s = 0.f, q = 0.f;
    for (int sl = 0; sl < 32; ++sl) {
        s += psum[((size_t)b * 32 + sl) * 128 + c];
        q += psq [((size_t)b * 32 + sl) * 128 + c];
    }
    const float cnt = (float)PER;
    float m   = s / cnt;
    float var = (q - cnt * m * m) / (cnt - 1.0f) + EPSV;   // unbiased + eps
    mean[b * 128 + c] = m;
    rstd[b * 128 + c] = 1.0f / sqrtf(var);
}

// norm + leaky -> bf16 activation (input of conv2)
__global__ void norm_leaky_bf16_kernel(
    const float* __restrict__ x,
    const float* __restrict__ mean, const float* __restrict__ rstd,
    const float* __restrict__ gamma, const float* __restrict__ beta,
    __bf16* __restrict__ out)
{
    const size_t g = (size_t)blockIdx.x * blockDim.x + threadIdx.x;
    const int c = (int)(g & 127);
    const int b = (int)(g >> 20);   // row = g>>7, batch = row>>13
    float v = (x[g] - mean[b * 128 + c]) * rstd[b * 128 + c] * gamma[c] + beta[c];
    v = (v >= 0.f) ? v : NEG_SLOPE * v;
    out[g] = (__bf16)v;
}

// norm + residual + leaky -> fp32 final output
__global__ void norm_res_leaky_kernel(
    const float* __restrict__ x,
    const float* __restrict__ mean, const float* __restrict__ rstd,
    const float* __restrict__ gamma, const float* __restrict__ beta,
    const float* __restrict__ feats, float* __restrict__ out)
{
    const size_t g = (size_t)blockIdx.x * blockDim.x + threadIdx.x;
    const int c = (int)(g & 127);
    const int b = (int)(g >> 20);
    float v = (x[g] - mean[b * 128 + c]) * rstd[b * 128 + c] * gamma[c] + beta[c];
    v += feats[g];
    out[g] = (v >= 0.f) ? v : NEG_SLOPE * v;
}

// ---------------------------------------------------------------------------
extern "C" void kernel_launch(void* const* d_in, const int* in_sizes, int n_in,
                              void* d_out, int out_size, void* d_ws, size_t ws_size,
                              hipStream_t stream) {
    (void)in_sizes; (void)n_in; (void)out_size; (void)ws_size;

    const float* feats  = (const float*)d_in[0];
    // d_in[1] = batch_ids (unused: batches are contiguous 8192-row blocks)
    const int*   nbr    = (const int*)  d_in[2];
    const float* w1     = (const float*)d_in[3];
    const float* gamma1 = (const float*)d_in[4];
    const float* beta1  = (const float*)d_in[5];
    const float* w2     = (const float*)d_in[6];
    const float* gamma2 = (const float*)d_in[7];
    const float* beta2  = (const float*)d_in[8];
    float* out = (float*)d_out;

    // workspace carve-up (256B aligned)
    char* ws = (char*)d_ws;
    size_t off = 0;
    auto carve = [&](size_t bytes) -> void* {
        void* p = ws + off;
        off += (bytes + 255) & ~(size_t)255;
        return p;
    };
    const size_t NC = (size_t)NPTS * CH;
    __bf16* fb    = (__bf16*)carve(NC * 2);                 // feats in bf16
    __bf16* act   = (__bf16*)carve(NC * 2);                 // act after norm1
    __bf16* pw1   = (__bf16*)carve((size_t)KVOL * CH * CH * 2);
    __bf16* pw2   = (__bf16*)carve((size_t)KVOL * CH * CH * 2);
    float*  convb = (float*)carve(NC * 4);                  // conv output (reused)
    float*  psum  = (float*)carve((size_t)NBATCH * 32 * 128 * 4);
    float*  psq   = (float*)carve((size_t)NBATCH * 32 * 128 * 4);
    float*  mean1 = (float*)carve(NBATCH * 128 * 4);
    float*  rstd1 = (float*)carve(NBATCH * 128 * 4);
    float*  mean2 = (float*)carve(NBATCH * 128 * 4);
    float*  rstd2 = (float*)carve(NBATCH * 128 * 4);

    const int RPK = KVOL * 4 * 8 * 512;  // packed weight elements per layer

    cvt_bf16_kernel<<<(int)(NC / 256), 256, 0, stream>>>(feats, fb, (int)NC);
    repack_w_kernel<<<(RPK + 255) / 256, 256, 0, stream>>>(w1, pw1);
    repack_w_kernel<<<(RPK + 255) / 256, 256, 0, stream>>>(w2, pw2);

    // conv1
    sparse_conv_wmma_kernel<<<NPTS / 256, 256, 0, stream>>>(fb, pw1, nbr, convb);
    stats_partial_kernel<<<NBATCH * 32, 256, 0, stream>>>(convb, psum, psq);
    stats_final_kernel<<<NBATCH, 128, 0, stream>>>(psum, psq, mean1, rstd1);
    norm_leaky_bf16_kernel<<<(int)(NC / 256), 256, 0, stream>>>(
        convb, mean1, rstd1, gamma1, beta1, act);

    // conv2
    sparse_conv_wmma_kernel<<<NPTS / 256, 256, 0, stream>>>(act, pw2, nbr, convb);
    stats_partial_kernel<<<NBATCH * 32, 256, 0, stream>>>(convb, psum, psq);
    stats_final_kernel<<<NBATCH, 128, 0, stream>>>(psum, psq, mean2, rstd2);
    norm_res_leaky_kernel<<<(int)(NC / 256), 256, 0, stream>>>(
        convb, mean2, rstd2, gamma2, beta2, feats, out);
}
